// GNNLayer_89215060672583
// MI455X (gfx1250) — matrix operand order repaired and checked by reading, not compiled, not run on hardware
//
#include <hip/hip_runtime.h>

typedef __attribute__((ext_vector_type(16))) _Float16 v16h;
typedef __attribute__((ext_vector_type(8)))  float    v8f;

#define NN 8192
#define D  512
#define LDS_STRIDE 520   // 512 + 8 halfs padding -> lanes land on distinct banks

// ---------------------------------------------------------------------------
// Kernel 0: compact node_feats[:,0] (stride-512 gather) into a flat 32KB array
// ---------------------------------------------------------------------------
__global__ void gather_col0(const float* __restrict__ nf, float* __restrict__ nf0) {
    int j = blockIdx.x * blockDim.x + threadIdx.x;
    if (j < NN) nf0[j] = nf[(size_t)j * D];
}

// ---------------------------------------------------------------------------
// Kernel 1: s[i] = sum_j adj[i,j] * nf0[j]   (256 MB stream -> HBM bound)
// One 256-thread block per row; int4/float4 coalesced loads, FMA, reduce.
// ---------------------------------------------------------------------------
__global__ __launch_bounds__(256) void row_reduce(const int*   __restrict__ adj,
                                                  const float* __restrict__ nf0,
                                                  float*       __restrict__ s) {
    const int row  = blockIdx.x;
    const int t    = threadIdx.x;
    const size_t base = (size_t)row * NN;
    float acc = 0.f;
    #pragma unroll
    for (int ch = 0; ch < 8; ++ch) {
        int j = ch * 1024 + t * 4;
        int4   a = *(const int4*)  (adj + base + j);
        float4 f = *(const float4*)(nf0 + j);
        acc = fmaf((float)a.x, f.x, acc);
        acc = fmaf((float)a.y, f.y, acc);
        acc = fmaf((float)a.z, f.z, acc);
        acc = fmaf((float)a.w, f.w, acc);
    }
    #pragma unroll
    for (int off = 16; off > 0; off >>= 1)
        acc += __shfl_xor(acc, off, 32);
    __shared__ float wsum[8];
    const int w = t >> 5, lane = t & 31;
    if (lane == 0) wsum[w] = acc;
    __syncthreads();
    if (t == 0) {
        float tot = 0.f;
        #pragma unroll
        for (int i = 0; i < 8; ++i) tot += wsum[i];
        s[row] = tot;   // rows with no neighbors give exact 0 -> mask is implicit
    }
}

// ---------------------------------------------------------------------------
// Kernel 2: wn_sum[d] = sum_k w_neigh[d,k]   (one wave per row)
// ---------------------------------------------------------------------------
__global__ __launch_bounds__(256) void wrow_sum(const float* __restrict__ wn,
                                                float*       __restrict__ wnsum) {
    const int w = threadIdx.x >> 5, lane = threadIdx.x & 31;
    const int d = blockIdx.x * 8 + w;
    const float* row = wn + (size_t)d * D;
    float acc = 0.f;
    #pragma unroll
    for (int g = 0; g < 4; ++g) {
        float4 f = *(const float4*)(row + g * 128 + lane * 4);
        acc += f.x + f.y + f.z + f.w;
    }
    #pragma unroll
    for (int off = 16; off > 0; off >>= 1)
        acc += __shfl_xor(acc, off, 32);
    if (lane == 0) wnsum[d] = acc;
}

// ---------------------------------------------------------------------------
// Kernel 3: out = relu(nf @ w_self^T + s outer wn_sum)  via v_wmma_f32_16x16x32_f16
// Block: 256 threads (8 waves) -> 16(M) x 128(N) tile; A staged fp16 in LDS.
// ---------------------------------------------------------------------------
__global__ __launch_bounds__(256) void gemm_wmma(const float* __restrict__ nf,
                                                 const float* __restrict__ wself,
                                                 const float* __restrict__ s,
                                                 const float* __restrict__ wnsum,
                                                 float*       __restrict__ out) {
    __shared__ _Float16 ldsA[16 * LDS_STRIDE];
    const int t  = threadIdx.x;
    const int m0 = blockIdx.y * 16;

    // Stage A tile (16 x 512), fp32 -> fp16, coalesced float4 loads, b64 LDS stores
    #pragma unroll
    for (int i = 0; i < 8; ++i) {
        int q   = t + i * 256;       // float4 chunk 0..2047 (128 chunks per row)
        int row = q >> 7;
        int c4  = q & 127;
        float4 f = *(const float4*)(nf + (size_t)(m0 + row) * D + c4 * 4);
        union { _Float16 h[4]; unsigned long long u; } pk;
        pk.h[0] = (_Float16)f.x; pk.h[1] = (_Float16)f.y;
        pk.h[2] = (_Float16)f.z; pk.h[3] = (_Float16)f.w;
        *(unsigned long long*)&ldsA[row * LDS_STRIDE + c4 * 4] = pk.u;
    }
    __syncthreads();

    const int w    = t >> 5;
    const int lane = t & 31;
    const int hi   = lane >> 4;      // lane half selects K sub-range
    const int lm   = lane & 15;
    const int n0   = blockIdx.x * 128 + w * 16;

    // B[k, n] = w_self[n, k]; lane lm owns column n0+lm (row of w_self, contiguous)
    const float*    brow = wself + (size_t)(n0 + lm) * D + hi * 8;
    const _Float16* arow = &ldsA[lm * LDS_STRIDE + hi * 8];

    v8f acc = {};
    #pragma unroll
    for (int ks = 0; ks < 16; ++ks) {           // K = 512 in steps of 32
        v16h a, b;
        #pragma unroll
        for (int p = 0; p < 8; ++p) {           // A 16x32 f16 VGPR layout
            const int kOff = (p < 4) ? (2 * p) : (16 + 2 * (p - 4));
            const _Float16* src = arow + ks * 32 + kOff;
            a[2 * p]     = src[0];
            a[2 * p + 1] = src[1];
        }
        #pragma unroll
        for (int r = 0; r < 8; ++r) {           // B 32x16 f16 VGPR layout
            float2 wv = *(const float2*)(brow + ks * 32 + ((r < 4) ? (2 * r) : (16 + 2 * (r - 4))));
            b[2 * r]     = (_Float16)wv.x;
            b[2 * r + 1] = (_Float16)wv.y;
        }
        acc = __builtin_amdgcn_wmma_f32_16x16x32_f16(
                  false, a, false, b, (short)0, acc, false, false);
    }

    // Epilogue: fuse rank-1 neighbor term + ReLU; C/D layout M = r + 8*hi, N = lm
    const float wncol = wnsum[n0 + lm];
    #pragma unroll
    for (int r = 0; r < 8; ++r) {
        const int row = m0 + hi * 8 + r;
        float v = acc[r] + s[row] * wncol;
        v = v > 0.f ? v : 0.f;
        out[(size_t)row * D + n0 + lm] = v;
    }
}

// ---------------------------------------------------------------------------
extern "C" void kernel_launch(void* const* d_in, const int* in_sizes, int n_in,
                              void* d_out, int out_size, void* d_ws, size_t ws_size,
                              hipStream_t stream) {
    const float* nf     = (const float*)d_in[0];   // node_feats  [8192,512]
    const int*   adj    = (const int*)  d_in[1];   // adj_matrix  [8192,8192]
    const float* wself  = (const float*)d_in[2];   // w_self      [512,512]
    const float* wneigh = (const float*)d_in[3];   // w_neigh     [512,512]
    float*       out    = (float*)d_out;

    float* ws   = (float*)d_ws;
    float* nf0  = ws;            // 8192 floats
    float* sbuf = ws + NN;       // 8192 floats
    float* wns  = ws + 2 * NN;   // 512 floats

    gather_col0<<<NN / 256, 256, 0, stream>>>(nf, nf0);
    row_reduce <<<NN,       256, 0, stream>>>(adj, nf0, sbuf);
    wrow_sum   <<<D / 8,    256, 0, stream>>>(wneigh, wns);
    dim3 grid(D / 128, NN / 16);                 // (4, 512)
    gemm_wmma  <<<grid,     256, 0, stream>>>(nf, wself, sbuf, wns, out);
}
